// Synthesis_455266533550
// MI455X (gfx1250) — matrix operand order repaired and checked by reading, not compiled, not run on hardware
//
#include <hip/hip_runtime.h>
#include <cstdint>
#include <cstddef>

// ---------------------------------------------------------------------------
// Types
// ---------------------------------------------------------------------------
typedef __bf16 bf16;
typedef __attribute__((ext_vector_type(16))) __bf16   v16bf;
typedef __attribute__((ext_vector_type(8)))  float    v8f;
typedef __attribute__((ext_vector_type(4)))  unsigned u32x4;
typedef int v4i __attribute__((vector_size(16)));   // matches builtin param type

#define AS1 __attribute__((address_space(1)))
#define AS3 __attribute__((address_space(3)))

#if __has_builtin(__builtin_amdgcn_global_load_async_to_lds_b128)
#define USE_ASYNC_LDS 1
#else
#define USE_ASYNC_LDS 0
#endif

__device__ __forceinline__ void wait_async0() {
#if __has_builtin(__builtin_amdgcn_s_wait_asynccnt)
  __builtin_amdgcn_s_wait_asynccnt(0);
#else
  asm volatile("s_wait_asynccnt 0x0" ::: "memory");
#endif
}

union Frag { v16bf v; u32x4 q[2]; };

// ---------------------------------------------------------------------------
// Weight prep: fp32 kernel [3,3,Cin,Cout] -> bf16 [Cout][K], K = 9*Cin,
// k = (ky*3+kx)*Cin + cin   (memory index of kernel is k*Cout + cout)
// ---------------------------------------------------------------------------
__global__ void wprep_kernel(const float* __restrict__ kern,
                             bf16* __restrict__ out, int K, int Cout) {
  int i = blockIdx.x * blockDim.x + threadIdx.x;
  if (i >= K * Cout) return;
  int k = i % K, co = i / K;
  out[(size_t)co * K + k] = (bf16)kern[(size_t)k * Cout + co];
}

// K2[cout] = sum over (kh,kw,cin) of kernel^2
__global__ void k2_kernel(const float* __restrict__ kern,
                          float* __restrict__ k2, int K, int Cout) {
  int co = blockIdx.x * blockDim.x + threadIdx.x;
  if (co >= Cout) return;
  float s = 0.f;
  for (int k = 0; k < K; ++k) {
    float v = kern[(size_t)k * Cout + co];
    s += v * v;
  }
  k2[co] = s;
}

// scale[b,cout] = (1+s) * rsqrt((1+s)^2 * K2 + eps),  s = w @ style_scale
__global__ void style_kernel(const float* __restrict__ w,
                             const float* __restrict__ ss,
                             const float* __restrict__ k2,
                             float* __restrict__ scale, int Cout) {
  int i = blockIdx.x * blockDim.x + threadIdx.x;
  if (i >= 4 * Cout) return;
  int b = i / Cout, j = i % Cout;
  float s = 0.f;
  for (int d = 0; d < 512; ++d) s += w[b * 512 + d] * ss[(size_t)d * Cout + j];
  float m = 1.f + s;
  scale[i] = m * rsqrtf(m * m * k2[j] + 1e-8f);
}

// ConstantInput: [1,512,4,4] fp32 -> act[b][hw][512] bf16, tiled over B=4
__global__ void const_kernel(const float* __restrict__ c, bf16* __restrict__ act) {
  int i = blockIdx.x * blockDim.x + threadIdx.x;
  if (i >= 4 * 16 * 512) return;
  int ch = i & 511, hw = (i >> 9) & 15, b = i >> 13;
  act[(((size_t)b * 16 + hw) << 9) + ch] = (bf16)c[ch * 16 + hw];
}

// ---------------------------------------------------------------------------
// Modulated 3x3 conv as implicit GEMM on WMMA bf16:
//   D[Cout, B*H*W] = W[Cout, 9*Cin] x im2col(X)[9*Cin, B*H*W]
// WG = 128 thr = 4 waves (2 M-waves x 2 N-waves); WG tile M=128, N=64;
// wave tile 64x32 = 4x2 fragments -> 8 v_wmma_f32_16x16x32_bf16 per K-step,
// A-frags reused across 2 N-frags, B-frags across 4 M-frags.
// A-tile (128x32 bf16 = 8KB) staged in LDS via async-to-LDS (one 64B row per
// thread per K-step); B-fragments gathered straight from global (activations
// are fully L2-resident: peak layer footprint ~70MB << 192MB L2).
// Epilogue fused: per-(b,cout) modulation scale, noise, bias, lrelu*sqrt(2).
// ---------------------------------------------------------------------------
__global__ __launch_bounds__(128) void conv3x3_mod_wmma(
    const bf16* __restrict__ actIn, const bf16* __restrict__ wgt,
    const float* __restrict__ scale, const float* __restrict__ noise,
    const float* __restrict__ noiseW, const float* __restrict__ bias,
    bf16* __restrict__ actOut,
    int Cin, int Cout, int H, int logH, int logCin, int up) {
  const int K     = 9 * Cin;
  const int HW    = H * H;
  const int logHW = 2 * logH;

  const int tid  = threadIdx.x;
  const int lane = tid & 31;
  const int wave = tid >> 5;
  const int wm   = wave >> 1;   // 0..1 (M): 64 rows each
  const int wn   = wave & 1;    // 0..1 (N): 32 cols each
  const int hi   = lane >> 4;   // lane half
  const int l16  = lane & 15;

  const int m0 = blockIdx.y * 128;
  const int n0 = blockIdx.x * 64;

  __shared__ __align__(16) bf16 smA[128 * 32];

  // Pixel decomposition for the two N-fragments (constant over K loop)
  int pb[2], poh[2], pow_[2];
#pragma unroll
  for (int nf = 0; nf < 2; ++nf) {
    int p   = n0 + wn * 32 + nf * 16 + l16;
    int b   = p >> logHW;
    int hw  = p & (HW - 1);
    pb[nf]  = b;
    poh[nf] = hw >> logH;
    pow_[nf] = hw & (H - 1);
  }
  const int Hin  = up ? (H >> 1) : H;
  const int HWin = Hin * Hin;

  v8f acc[4][2] = {};

  for (int k0 = 0; k0 < K; k0 += 32) {
    __syncthreads();  // protect smA from readers of previous iteration
    {
      // Each of the 128 threads stages one full 32-K (64B) weight row.
      const bf16* g = wgt + (size_t)(m0 + tid) * K + k0;
      bf16* l = &smA[tid * 32];
#if USE_ASYNC_LDS
      __builtin_amdgcn_global_load_async_to_lds_b128((AS1 v4i*)(g),      (AS3 v4i*)(l),      0, 0);
      __builtin_amdgcn_global_load_async_to_lds_b128((AS1 v4i*)(g + 8),  (AS3 v4i*)(l + 8),  0, 0);
      __builtin_amdgcn_global_load_async_to_lds_b128((AS1 v4i*)(g + 16), (AS3 v4i*)(l + 16), 0, 0);
      __builtin_amdgcn_global_load_async_to_lds_b128((AS1 v4i*)(g + 24), (AS3 v4i*)(l + 24), 0, 0);
      wait_async0();
#else
      const u32x4* gq = (const u32x4*)g;
      u32x4* lq = (u32x4*)l;
      lq[0] = gq[0]; lq[1] = gq[1]; lq[2] = gq[2]; lq[3] = gq[3];
#endif
    }
    __syncthreads();

    // A fragments from LDS (16-bit A layout: lanes0-15 K0..7|K16..23,
    // lanes16-31 K8..15|K24..31, row M = lane&15)
    Frag a[4];
#pragma unroll
    for (int mf = 0; mf < 4; ++mf) {
      int r = wm * 64 + mf * 16 + l16;
      const bf16* base = &smA[r * 32 + hi * 8];
      a[mf].q[0] = *(const u32x4*)(base);
      a[mf].q[1] = *(const u32x4*)(base + 16);
    }

    // B fragments from global with fused im2col + nearest-2x upsample.
    // 32-K chunk lies inside one conv tap (Cin multiple of 32).
    int tap  = k0 >> logCin;
    int cinb = k0 & (Cin - 1);
    int ky   = tap / 3, kx = tap % 3;

    Frag bfr[2];
#pragma unroll
    for (int nf = 0; nf < 2; ++nf) {
      int ih = poh[nf] + ky - 1;
      int iw = pow_[nf] + kx - 1;
      bool ok = ((unsigned)ih < (unsigned)H) && ((unsigned)iw < (unsigned)H);
      if (ok) {
        int sh = up ? (ih >> 1) : ih;
        int sw = up ? (iw >> 1) : iw;
        size_t idx = ((size_t)(pb[nf] * HWin + sh * Hin + sw) << logCin)
                     + cinb + hi * 8;
        const u32x4* q = (const u32x4*)(actIn + idx);
        bfr[nf].q[0] = q[0];
        bfr[nf].q[1] = q[2];
        __builtin_prefetch(actIn + idx + (size_t)Cin, 0, 1);
      } else {
        bfr[nf].q[0] = (u32x4)0u;  // SAME padding
        bfr[nf].q[1] = (u32x4)0u;
      }
    }

#pragma unroll
    for (int mf = 0; mf < 4; ++mf)
#pragma unroll
      for (int nf = 0; nf < 2; ++nf)
        acc[mf][nf] = __builtin_amdgcn_wmma_f32_16x16x32_bf16(
            false, a[mf].v, false, bfr[nf].v, (short)0, acc[mf][nf],
            false, false);
  }

  // Fused epilogue: modulation scale, noise, bias, leaky-relu * sqrt(2).
  // C layout: lanes0-15 -> M = r, lanes16-31 -> M = 8+r, N = lane&15.
  const float nw = noiseW[0];
#pragma unroll
  for (int nf = 0; nf < 2; ++nf) {
    int p  = n0 + wn * 32 + nf * 16 + l16;
    int b  = p >> logHW;
    int hw = p & (HW - 1);
    float nz = nw * noise[(size_t)b * HW + hw];
    bf16* outp = actOut + ((size_t)b * HW + hw) * Cout;
#pragma unroll
    for (int mf = 0; mf < 4; ++mf) {
      int co0 = m0 + wm * 64 + mf * 16 + hi * 8;
      union { u32x4 q; bf16 h[8]; } st;
#pragma unroll
      for (int r = 0; r < 8; ++r) {
        float v = acc[mf][nf][r] * scale[(size_t)b * Cout + co0 + r] + nz +
                  bias[co0 + r];
        v = (v < 0.f ? 0.2f * v : v) * 1.4142135623730951f;
        st.h[r] = (bf16)v;
      }
      *(u32x4*)(outp + co0) = st.q;
    }
  }
}

// ---------------------------------------------------------------------------
// ToRGB: 1x1 lr-equalized conv Cin->3 (+ nearest-2x skip accumulate)
// ---------------------------------------------------------------------------
__global__ void torgb_kernel(const bf16* __restrict__ act,
                             const float* __restrict__ k,
                             const float* __restrict__ bias,
                             const float* __restrict__ skipIn,
                             float* __restrict__ out, int C, int H,
                             int hasSkip) {
  int i = blockIdx.x * blockDim.x + threadIdx.x;
  int HW = H * H;
  if (i >= 4 * 3 * HW) return;
  int x = i % H;
  int y = (i / H) % H;
  int o = (i / HW) % 3;
  int b = i / (3 * HW);
  const bf16* a = act + ((size_t)b * HW + (size_t)y * H + x) * C;
  float s = 0.f;
  for (int c = 0; c < C; ++c) s += (float)a[c] * k[o * C + c];
  s = s * rsqrtf((float)C) + bias[o];
  if (hasSkip) {
    int Hh = H >> 1;
    s += skipIn[((size_t)(b * 3 + o) * Hh + (y >> 1)) * Hh + (x >> 1)];
  }
  out[((size_t)(b * 3 + o) * H + y) * H + x] = s;
}

// ---------------------------------------------------------------------------
// Host orchestration
// ---------------------------------------------------------------------------
static inline int ilog2(int v) {
  int l = 0;
  while ((1 << l) < v) ++l;
  return l;
}

extern "C" void kernel_launch(void* const* d_in, const int* in_sizes, int n_in,
                              void* d_out, int out_size, void* d_ws,
                              size_t ws_size, hipStream_t stream) {
  (void)in_sizes; (void)n_in; (void)out_size; (void)ws_size;
  const int NL = 13;
  static const int cin_[NL]  = {512,512,512,512,512,512,512,512,512,512,256,256,128};
  static const int cout_[NL] = {512,512,512,512,512,512,512,512,512,256,256,128,128};
  static const int res_[NL]  = {4,  8,  8,  16, 16, 32, 32, 64, 64, 128,128,256,256};
  static const int up_[NL]   = {0,  1,  0,  1,  0,  1,  0,  1,  0,  1,  0,  1,  0};

  const float* w        = (const float*)d_in[0];
  const float* constant = (const float*)d_in[1];
  const float *convK[NL], *convSS[NL], *convNW[NL], *convB[NL], *noise[NL];
  for (int i = 0; i < NL; ++i) {
    convK[i]  = (const float*)d_in[2 + i];
    convSS[i] = (const float*)d_in[15 + i];
    convNW[i] = (const float*)d_in[28 + i];
    convB[i]  = (const float*)d_in[41 + i];
    noise[i]  = (const float*)d_in[68 + i];
  }
  const float *rgbK[7], *rgbB[7];
  for (int s = 0; s < 7; ++s) {
    rgbK[s] = (const float*)d_in[54 + s];
    rgbB[s] = (const float*)d_in[61 + s];
  }

  // Workspace carve
  char* ws = (char*)d_ws;
  size_t off = 0;
  auto carve = [&](size_t bytes) -> char* {
    char* p = ws + off;
    off += (bytes + 255) & ~(size_t)255;
    return p;
  };
  const size_t ACT_ELEMS = (size_t)4 * 256 * 256 * 128;  // peak activation
  bf16*  actA  = (bf16*)carve(ACT_ELEMS * 2);
  bf16*  actB  = (bf16*)carve(ACT_ELEMS * 2);
  bf16*  wbuf  = (bf16*)carve((size_t)512 * 9 * 512 * 2);
  float* k2b   = (float*)carve(512 * 4);
  float* scb   = (float*)carve(4 * 512 * 4);
  float* sbuf0 = (float*)carve((size_t)4 * 3 * 128 * 128 * 4);
  float* sbuf1 = (float*)carve((size_t)4 * 3 * 128 * 128 * 4);
  float* sbuf[2] = {sbuf0, sbuf1};

  const_kernel<<<(4 * 16 * 512 + 255) / 256, 256, 0, stream>>>(constant, actA);

  bf16* cur = actA;
  bf16* nxt = actB;
  float* prevSkip = nullptr;
  int sidx = 0, rgbIdx = 0;

  for (int i = 0; i < NL; ++i) {
    const int Cin = cin_[i], Cout = cout_[i], H = res_[i];
    const int K = 9 * Cin;
    const int logH = ilog2(H), logCin = ilog2(Cin);

    wprep_kernel<<<(K * Cout + 255) / 256, 256, 0, stream>>>(convK[i], wbuf, K, Cout);
    k2_kernel<<<(Cout + 255) / 256, 256, 0, stream>>>(convK[i], k2b, K, Cout);
    style_kernel<<<(4 * Cout + 255) / 256, 256, 0, stream>>>(w, convSS[i], k2b, scb, Cout);

    const int N = 4 * H * H;
    dim3 grid(N / 64, Cout / 128);
    conv3x3_mod_wmma<<<grid, 128, 0, stream>>>(cur, wbuf, scb, noise[i],
                                               convNW[i], convB[i], nxt,
                                               Cin, Cout, H, logH, logCin,
                                               up_[i]);
    bf16* t = cur; cur = nxt; nxt = t;

    if ((i & 1) == 0) {  // ToRGB after L0, L2, ..., L12
      bool last = (i == NL - 1);
      float* target = last ? (float*)d_out : sbuf[sidx];
      torgb_kernel<<<(4 * 3 * H * H + 255) / 256, 256, 0, stream>>>(
          cur, rgbK[rgbIdx], rgbB[rgbIdx], prevSkip, target, Cout, H,
          prevSkip != nullptr);
      prevSkip = target;
      sidx ^= 1;
      ++rgbIdx;
    }
  }
}
